// Block_63934883168776
// MI455X (gfx1250) — compile-verified, matrix operand order checked by reading, not run on hardware
//
#include <hip/hip_runtime.h>

// ---------------- problem constants ----------------
constexpr int Bc  = 4;
constexpr int Sc  = 2048;
constexpr int Dc  = 1024;
constexpr int Hc  = 16;
constexpr int HDc = 64;        // head dim
constexpr int DFFc = 4096;
constexpr int Mrows = Bc * Sc; // 8192

typedef __attribute__((ext_vector_type(16))) _Float16 v16h;
typedef __attribute__((ext_vector_type(8)))  _Float16 v8h;
typedef __attribute__((ext_vector_type(8)))  float    v8f;

__device__ __forceinline__ v8f zero8() {
  v8f z;
#pragma unroll
  for (int i = 0; i < 8; ++i) z[i] = 0.f;
  return z;
}

__device__ __forceinline__ v8f wmma_f16(v16h a, v16h b, v8f c) {
  // D = A(16x32 f16) * B(32x16 f16) + C(16x16 f32)
  return __builtin_amdgcn_wmma_f32_16x16x32_f16(false, a, false, b,
                                                (short)0, c, false, false);
}

// Async global -> LDS 16-byte copy (CDNA5, ASYNCcnt).
// lds_addr: LDS byte address (low 32 bits of the generic shared pointer).
__device__ __forceinline__ void async_b128(unsigned lds_addr, const void* gaddr) {
  asm volatile("global_load_async_to_lds_b128 %0, %1, off"
               :: "v"(lds_addr), "v"(gaddr) : "memory");
}
__device__ __forceinline__ void wait_async0() {
  asm volatile("s_wait_asynccnt 0x0" ::: "memory");
}
__device__ __forceinline__ unsigned lds_addr_of(const void* p) {
  return (unsigned)(unsigned long long)p;
}

// Load one 16x32 A-style fragment (B-style symmetric: col<->row swap).
// rowptr: start of this lane's row (row = lane&15).  h = lane>>4.
// element j <-> K = (j>>3)*16 + h*8 + (j&7)  => two contiguous 16B loads.
__device__ __forceinline__ v16h load_frag(const _Float16* rowptr, int k, int h) {
  v8h lo = *(const v8h*)(rowptr + k + h * 8);
  v8h hi = *(const v8h*)(rowptr + k + 16 + h * 8);
  v16h r;
#pragma unroll
  for (int i = 0; i < 8; ++i) { r[i] = lo[i]; r[i + 8] = hi[i]; }
  return r;
}

// ---------------- weight transpose + f32->f16 convert ----------------
// W: [K][N] f32  ->  Wt: [N][K] f16   (block 32x8, 32x32 tiles via LDS)
__global__ void wt_to_f16_t(const float* __restrict__ W, _Float16* __restrict__ Wt,
                            int K, int N) {
  __shared__ float tile[32][33];
  int kb = blockIdx.y * 32, nb = blockIdx.x * 32;
  int tx = threadIdx.x, ty = threadIdx.y;
#pragma unroll
  for (int i = ty; i < 32; i += 8)
    tile[i][tx] = W[(long)(kb + i) * N + nb + tx];
  __syncthreads();
#pragma unroll
  for (int i = ty; i < 32; i += 8)
    Wt[(long)(nb + i) * K + kb + tx] = (_Float16)tile[tx][i];
}

// ---------------- fused LayerNorm -> f16 ----------------
__global__ void __launch_bounds__(256) ln_to_f16(const float* __restrict__ x,
                                                 const float* __restrict__ g,
                                                 const float* __restrict__ bb,
                                                 _Float16* __restrict__ out) {
  __shared__ float red[256];
  long row = blockIdx.x;
  const float* xr = x + row * Dc;
  int t = threadIdx.x, base = t * 4;
  float v[4];
  float s = 0.f;
#pragma unroll
  for (int i = 0; i < 4; ++i) { v[i] = xr[base + i]; s += v[i]; }
  red[t] = s; __syncthreads();
  for (int off = 128; off > 0; off >>= 1) {
    if (t < off) red[t] += red[t + off];
    __syncthreads();
  }
  float mean = red[0] * (1.f / Dc);
  __syncthreads();
  float sq = 0.f;
#pragma unroll
  for (int i = 0; i < 4; ++i) { float d = v[i] - mean; sq += d * d; }
  red[t] = sq; __syncthreads();
  for (int off = 128; off > 0; off >>= 1) {
    if (t < off) red[t] += red[t + off];
    __syncthreads();
  }
  float inv = rsqrtf(red[0] * (1.f / Dc) + 1e-5f);
  _Float16* o = out + row * Dc;
#pragma unroll
  for (int i = 0; i < 4; ++i)
    o[base + i] = (_Float16)((v[i] - mean) * inv * g[base + i] + bb[base + i]);
}

// ---------------- WMMA GEMM: C[M,N] = A[M,K] * Wt[N,K]^T ----------------
// Block tile 64 x 256, 8 waves (2x4), wave tile 32x64 (2x4 fragments).
// K staged in 32-wide double-buffered LDS tiles via async global->LDS copies.
// EPI 0: +bias -> f16    EPI 1: +bias, gelu -> f16    EPI 2: +bias +resid -> f32
template <int EPI>
__global__ void __launch_bounds__(256) gemm_wmma(const _Float16* __restrict__ A,
                                                 const _Float16* __restrict__ Wt,
                                                 const float* __restrict__ bias,
                                                 const float* __restrict__ resid,
                                                 float* __restrict__ outF,
                                                 _Float16* __restrict__ outH,
                                                 int M, int N, int K) {
  __shared__ __align__(16) _Float16 Asm[2][64 * 32];    // [row][k]
  __shared__ __align__(16) _Float16 Bsm[2][256 * 32];   // [col][k]

  int tid  = threadIdx.x;
  int wave = tid >> 5;
  int lane = tid & 31;
  int h    = lane >> 4;
  int c    = lane & 15;
  int wr = wave >> 2, wc = wave & 3;            // 2 x 4 waves
  int m0b = blockIdx.y * 64;
  int n0b = blockIdx.x * 256;
  int m0w = wr * 32;
  int n0w = wc * 64;

  // staging: 16B chunk tid covers row = tid>>2, halfs (tid&3)*8..+7
  // => LDS byte offset tid*16.  B tile has 256 rows -> 4 chunks per thread.
  const _Float16* agp  = A  + (long)(m0b + (tid >> 2)) * K + (tid & 3) * 8;
  const _Float16* bgp0 = Wt + (long)(n0b +       (tid >> 2)) * K + (tid & 3) * 8;
  const _Float16* bgp1 = Wt + (long)(n0b +  64 + (tid >> 2)) * K + (tid & 3) * 8;
  const _Float16* bgp2 = Wt + (long)(n0b + 128 + (tid >> 2)) * K + (tid & 3) * 8;
  const _Float16* bgp3 = Wt + (long)(n0b + 192 + (tid >> 2)) * K + (tid & 3) * 8;
  unsigned aL0 = lds_addr_of(&Asm[0][0]) + tid * 16;
  unsigned aL1 = lds_addr_of(&Asm[1][0]) + tid * 16;
  unsigned bL0 = lds_addr_of(&Bsm[0][0]) + tid * 16;
  unsigned bL1 = lds_addr_of(&Bsm[1][0]) + tid * 16;

  v8f acc[2][4];
#pragma unroll
  for (int i = 0; i < 2; ++i)
#pragma unroll
    for (int j = 0; j < 4; ++j) acc[i][j] = zero8();

  auto stage = [&](int p, int k) {
    unsigned aL = p ? aL1 : aL0;
    unsigned bL = p ? bL1 : bL0;
    async_b128(aL, agp + k);
    async_b128(bL, bgp0 + k);
    async_b128(bL + 4096, bgp1 + k);
    async_b128(bL + 8192, bgp2 + k);
    async_b128(bL + 12288, bgp3 + k);
  };

  stage(0, 0);
  int p = 0;
  for (int k = 0; k < K; k += 32) {
    wait_async0();
    __syncthreads();
    if (k + 32 < K) stage(p ^ 1, k + 32);

    const _Float16* Ap = &Asm[p][(m0w + c) * 32];
    v16h a0 = load_frag(Ap, 0, h);
    v16h a1 = load_frag(Ap + 16 * 32, 0, h);
    v16h bf[4];
#pragma unroll
    for (int j = 0; j < 4; ++j)
      bf[j] = load_frag(&Bsm[p][(n0w + j * 16 + c) * 32], 0, h);
#pragma unroll
    for (int j = 0; j < 4; ++j) {
      acc[0][j] = wmma_f16(a0, bf[j], acc[0][j]);
      acc[1][j] = wmma_f16(a1, bf[j], acc[1][j]);
    }
    __syncthreads();
    p ^= 1;
  }

#pragma unroll
  for (int i = 0; i < 2; ++i) {
#pragma unroll
    for (int j = 0; j < 4; ++j) {
      int coln = n0b + n0w + j * 16 + c;
      float bv = bias[coln];
#pragma unroll
      for (int r = 0; r < 8; ++r) {
        long row = (long)m0b + m0w + i * 16 + r + 8 * h;
        float v = acc[i][j][r] + bv;
        if (EPI == 0) {
          outH[row * N + coln] = (_Float16)v;
        } else if (EPI == 1) {
          float t3 = v * v * v;
          float gl = 0.5f * v * (1.f + tanhf(0.7978845608f * (v + 0.044715f * t3)));
          outH[row * N + coln] = (_Float16)gl;
        } else {
          outF[row * N + coln] = v + resid[row * N + coln];
        }
      }
    }
  }
}

// ---------------- flash attention ----------------
// grid: (S/128, H, B), block 256 (8 waves); each wave = one 16-row Q tile.
__global__ void __launch_bounds__(256) attn_wmma(const _Float16* __restrict__ qkv,
                                                 const float* __restrict__ am,
                                                 _Float16* __restrict__ aout) {
  __shared__ __align__(16) _Float16 Kt[32 * 64];   // [key][hd]
  __shared__ __align__(16) _Float16 Vt[64 * 32];   // [hd][key]  (transposed)
  __shared__ __align__(16) _Float16 Pb[8 * 16 * 32];

  int tid  = threadIdx.x;
  int wave = tid >> 5;
  int lane = tid & 31;
  int h5   = lane >> 4;
  int c    = lane & 15;
  int b    = blockIdx.z;
  int head = blockIdx.y;
  int qbase = blockIdx.x * 128 + wave * 16;

  _Float16* Pw = &Pb[wave * 512];
  unsigned KtL = lds_addr_of(&Kt[0]) + tid * 16;

  const _Float16* qrowp = qkv + ((long)(b * Sc + qbase + c)) * (3 * Dc) + head * HDc;
  v16h q0 = load_frag(qrowp, 0, h5);
  v16h q1 = load_frag(qrowp, 32, h5);

  float mrow[8], lrow[8];
  v8f o[4];
#pragma unroll
  for (int r = 0; r < 8; ++r) { mrow[r] = -1e30f; lrow[r] = 0.f; }
#pragma unroll
  for (int t = 0; t < 4; ++t) o[t] = zero8();

  int nkt = blockIdx.x * 4 + 4;        // uniform causal bound for the block
  const float* amr = am + (long)b * Sc;

  for (int kt = 0; kt < nkt; ++kt) {
    int kb0 = kt * 32;
    __syncthreads();
    // stage K tile [32 keys x 64 hd] via async copy (LDS offset == tid*16)
    {
      const _Float16* ksrc =
          qkv + ((long)(b * Sc + kb0 + (tid >> 3))) * (3 * Dc) + Dc + head * HDc +
          (tid & 7) * 8;
      async_b128(KtL, ksrc);
      // stage V transposed: Vt[hd][key] (element permute -> regular ds stores)
      int vkey = tid >> 3, hd0 = (tid & 7) * 8;
      const _Float16* vsrc =
          qkv + ((long)(b * Sc + kb0 + vkey)) * (3 * Dc) + 2 * Dc + head * HDc + hd0;
      v8h vv = *(const v8h*)vsrc;
#pragma unroll
      for (int i = 0; i < 8; ++i) Vt[(hd0 + i) * 32 + vkey] = vv[i];
      wait_async0();
    }
    __syncthreads();

    // scores: 16 q-rows x 32 keys via 4 WMMAs
    v16h kb00 = load_frag(&Kt[c * 64], 0, h5);
    v16h kb01 = load_frag(&Kt[c * 64], 32, h5);
    v16h kb10 = load_frag(&Kt[(16 + c) * 64], 0, h5);
    v16h kb11 = load_frag(&Kt[(16 + c) * 64], 32, h5);
    v8f s0 = zero8(), s1 = zero8();
    s0 = wmma_f16(q0, kb00, s0); s0 = wmma_f16(q1, kb01, s0);
    s1 = wmma_f16(q0, kb10, s1); s1 = wmma_f16(q1, kb11, s1);

    int key0 = kb0 + c, key1 = kb0 + 16 + c;
    float am0 = amr[key0], am1 = amr[key1];
#pragma unroll
    for (int r = 0; r < 8; ++r) {
      int gq = qbase + r + 8 * h5;
      float v0 = s0[r] * 0.125f; if (key0 > gq) v0 = -10000.f; v0 += am0;
      float v1 = s1[r] * 0.125f; if (key1 > gq) v1 = -10000.f; v1 += am1;
      s0[r] = v0; s1[r] = v1;
    }

    // online softmax: reduce across the 16 lanes holding one row
#pragma unroll
    for (int r = 0; r < 8; ++r) {
      float mx = fmaxf(s0[r], s1[r]);
      mx = fmaxf(mx, __shfl_xor(mx, 1));
      mx = fmaxf(mx, __shfl_xor(mx, 2));
      mx = fmaxf(mx, __shfl_xor(mx, 4));
      mx = fmaxf(mx, __shfl_xor(mx, 8));
      float mnew  = fmaxf(mrow[r], mx);
      float alpha = __expf(mrow[r] - mnew);
      float p0 = __expf(s0[r] - mnew);
      float p1 = __expf(s1[r] - mnew);
      float ps = p0 + p1;
      ps += __shfl_xor(ps, 1);
      ps += __shfl_xor(ps, 2);
      ps += __shfl_xor(ps, 4);
      ps += __shfl_xor(ps, 8);
      lrow[r] = lrow[r] * alpha + ps;
      mrow[r] = mnew;
#pragma unroll
      for (int t = 0; t < 4; ++t) o[t][r] *= alpha;
      Pw[(r + 8 * h5) * 32 + c] = (_Float16)p0;
      Pw[(r + 8 * h5) * 32 + 16 + c] = (_Float16)p1;
    }
    asm volatile("" ::: "memory");    // DS ops are in-order per wave

    // O += P(16x32) * V(32x64) : 4 WMMAs
    v16h pf = load_frag(&Pw[c * 32], 0, h5);
#pragma unroll
    for (int t = 0; t < 4; ++t) {
      v16h vb = load_frag(&Vt[(t * 16 + c) * 32], 0, h5);
      o[t] = wmma_f16(pf, vb, o[t]);
    }
  }

  // normalize + store f16 context
#pragma unroll
  for (int t = 0; t < 4; ++t)
#pragma unroll
    for (int r = 0; r < 8; ++r) {
      int gq = qbase + r + 8 * h5;
      float val = o[t][r] / lrow[r];
      aout[((long)(b * Sc + gq)) * Dc + head * HDc + t * 16 + c] = (_Float16)val;
    }
}

// ---------------- launch ----------------
extern "C" void kernel_launch(void* const* d_in, const int* in_sizes, int n_in,
                              void* d_out, int out_size, void* d_ws, size_t ws_size,
                              hipStream_t stream) {
  const float* hid   = (const float*)d_in[0];
  const float* amask = (const float*)d_in[1];
  const float* ln1g  = (const float*)d_in[2];
  const float* ln1b  = (const float*)d_in[3];
  const float* Wattn = (const float*)d_in[4];
  const float* battn = (const float*)d_in[5];
  const float* Wproj = (const float*)d_in[6];
  const float* bproj = (const float*)d_in[7];
  const float* ln2g  = (const float*)d_in[8];
  const float* ln2b  = (const float*)d_in[9];
  const float* Wfc   = (const float*)d_in[10];
  const float* bfc   = (const float*)d_in[11];
  const float* Wfc2  = (const float*)d_in[12];
  const float* bfc2  = (const float*)d_in[13];

  char* ws = (char*)d_ws;
  size_t off = 0;
  auto alloc = [&](size_t bytes) {
    char* p = ws + off;
    off += (bytes + 255) & ~(size_t)255;
    return p;
  };
  _Float16* act16   = (_Float16*)alloc((size_t)Mrows * Dc * 2);   // h16 / a16 / h2ln16
  _Float16* qkv16   = (_Float16*)alloc((size_t)Mrows * 3 * Dc * 2);
  float*    x1      = (float*)  alloc((size_t)Mrows * Dc * 4);
  _Float16* g16     = (_Float16*)alloc((size_t)Mrows * DFFc * 2);
  _Float16* wattn_t = (_Float16*)alloc((size_t)3 * Dc * Dc * 2);
  _Float16* wproj_t = (_Float16*)alloc((size_t)Dc * Dc * 2);
  _Float16* wfc_t   = (_Float16*)alloc((size_t)Dc * DFFc * 2);
  _Float16* wfc2_t  = (_Float16*)alloc((size_t)DFFc * Dc * 2);

  dim3 tb(32, 8);
  wt_to_f16_t<<<dim3(3 * Dc / 32, Dc / 32), tb, 0, stream>>>(Wattn, wattn_t, Dc, 3 * Dc);
  wt_to_f16_t<<<dim3(Dc / 32, Dc / 32),     tb, 0, stream>>>(Wproj, wproj_t, Dc, Dc);
  wt_to_f16_t<<<dim3(DFFc / 32, Dc / 32),   tb, 0, stream>>>(Wfc,   wfc_t,   Dc, DFFc);
  wt_to_f16_t<<<dim3(Dc / 32, DFFc / 32),   tb, 0, stream>>>(Wfc2,  wfc2_t,  DFFc, Dc);

  // LN1 -> f16
  ln_to_f16<<<Mrows, 256, 0, stream>>>(hid, ln1g, ln1b, act16);
  // qkv = h @ W_attn + b_attn (f16 out)
  gemm_wmma<0><<<dim3(3 * Dc / 256, Mrows / 64), 256, 0, stream>>>(
      act16, wattn_t, battn, nullptr, nullptr, qkv16, Mrows, 3 * Dc, Dc);
  // attention -> act16 (context tensor)
  attn_wmma<<<dim3(Sc / 128, Hc, Bc), 256, 0, stream>>>(qkv16, amask, act16);
  // x1 = a @ W_proj + b_proj + x (f32)
  gemm_wmma<2><<<dim3(Dc / 256, Mrows / 64), 256, 0, stream>>>(
      act16, wproj_t, bproj, hid, x1, nullptr, Mrows, Dc, Dc);
  // LN2 -> f16
  ln_to_f16<<<Mrows, 256, 0, stream>>>(x1, ln2g, ln2b, act16);
  // g = gelu(h2 @ W_fc + b_fc) (f16)
  gemm_wmma<1><<<dim3(DFFc / 256, Mrows / 64), 256, 0, stream>>>(
      act16, wfc_t, bfc, nullptr, nullptr, g16, Mrows, DFFc, Dc);
  // out = g @ W_fc2 + b_fc2 + x1 (f32)
  gemm_wmma<2><<<dim3(Dc / 256, Mrows / 64), 256, 0, stream>>>(
      g16, wfc2_t, bfc2, x1, (float*)d_out, nullptr, Mrows, Dc, DFFc);
}
